// MHSAWithMLPBlock_89163521065532
// MI455X (gfx1250) — matrix operand order repaired and checked by reading, not compiled, not run on hardware
//
#include <hip/hip_runtime.h>
#include <hip/hip_bf16.h>

typedef __attribute__((ext_vector_type(16))) _Float16 v16h;
typedef __attribute__((ext_vector_type(8)))  float    v8f;

union H2 { unsigned int u; _Float16 h[2]; };

#define Bc 4
#define Cc 128
#define Nn 4096            // H*W
#define HEADSc 4
#define HDc 32
#define HIDDENc 256
#define NTOK (Bc*Nn)       // 16384

static __device__ __forceinline__ v8f wmma16(const v16h& a, const v16h& b, const v8f& c) {
    return __builtin_amdgcn_wmma_f32_16x16x32_f16(false, a, false, b, (short)0, c, false, false);
}

// ---------------------------------------------------------------------------
// Kernel 0: one-shot f32 -> f16 weight conversion (weights stay L2-resident)
// ---------------------------------------------------------------------------
__global__ __launch_bounds__(256) void k_cvt(const float* __restrict__ a,
                                             _Float16* __restrict__ b, int n) {
    int i = blockIdx.x*256 + threadIdx.x;
    if (i < n) b[i] = (_Float16)a[i];
}

// ---------------------------------------------------------------------------
// Kernel 1: LN1 (over channels) + QKV 1x1-conv GEMM  [16384 x 128] x [128 x 384]
// Q,K -> f16 [b*HEADS+h][n][32] ; V -> f16 transposed [b*HEADS+h][32][n]
// ---------------------------------------------------------------------------
__global__ __launch_bounds__(256) void k_ln_qkv(
    const float* __restrict__ x, const float* __restrict__ ln1w, const float* __restrict__ ln1b,
    const _Float16* __restrict__ qkvw,
    _Float16* __restrict__ qws, _Float16* __restrict__ kws, _Float16* __restrict__ vtws)
{
    __shared__ _Float16 sx[128][128];           // 32 KB: 128 tokens x 128 channels (f16)
    const int tid = threadIdx.x, blk = blockIdx.x;

    if (tid < 128) {                            // LN phase: 1 thread per token, coalesced over hw
        int token = blk*128 + tid;
        int b = token >> 12, hw = token & 4095;
        const float* xb = x + (size_t)b*Cc*Nn + hw;
        float s = 0.f, ss = 0.f;
        for (int c = 0; c < Cc; ++c) { float v = xb[(size_t)c*Nn]; s += v; ss += v*v; }
        float mu = s * (1.f/128.f);
        float var = ss * (1.f/128.f) - mu*mu;
        float rstd = rsqrtf(var + 1e-5f);
        for (int c = 0; c < Cc; ++c)
            sx[tid][c] = (_Float16)((xb[(size_t)c*Nn] - mu)*rstd*ln1w[c] + ln1b[c]);
    }
    __syncthreads();

    const int wave = tid >> 5, lane = tid & 31;
    const int half = lane >> 4, lo = lane & 15;
    const int rt = wave * 16;                   // 8 waves x 16 token-rows = 128

    v16h afr[4];                                // A fragments, 4 K-steps of 32
    const unsigned int* srow = (const unsigned int*)&sx[rt + lo][0];
    #pragma unroll
    for (int s = 0; s < 4; ++s) {
        #pragma unroll
        for (int j = 0; j < 8; ++j) {
            int k0 = 32*s + 16*(j>>2) + 8*half + 2*(j&3);
            H2 t; t.u = srow[k0 >> 1];
            afr[s][2*j] = t.h[0]; afr[s][2*j+1] = t.h[1];
        }
    }

    for (int nt = 0; nt < 24; ++nt) {           // 384 outputs / 16
        int n = nt*16 + lo;
        v16h bf[4];                             // prefetch ALL K-step B-frags (one load clause)
        #pragma unroll
        for (int s = 0; s < 4; ++s) {
            #pragma unroll
            for (int j = 0; j < 8; ++j) {
                int k0 = 32*s + 16*(j>>2) + 8*half + 2*(j&3);
                H2 t; t.u = *(const unsigned int*)(qkvw + (size_t)n*Cc + k0);
                bf[s][2*j] = t.h[0]; bf[s][2*j+1] = t.h[1];
            }
        }
        v8f acc = {};
        #pragma unroll
        for (int s = 0; s < 4; ++s) acc = wmma16(afr[s], bf[s], acc);

        if (nt < 16) {                          // Q / K: row-major [bh][n][32]
            _Float16* dst = (nt < 8) ? qws : kws;
            int o = n & 127, head = o >> 5, hd = o & 31;
            #pragma unroll
            for (int r = 0; r < 8; ++r) {
                int token = blk*128 + rt + r + 8*half;
                int b = token >> 12, ntok = token & 4095;
                dst[(((size_t)(b*HEADSc + head))*Nn + ntok)*HDc + hd] = (_Float16)acc[r];
            }
        } else {                                // V: transposed [bh][32][n]
            int o = n - 256, head = o >> 5, hd = o & 31;
            #pragma unroll
            for (int r = 0; r < 8; ++r) {
                int token = blk*128 + rt + r + 8*half;
                int b = token >> 12, ntok = token & 4095;
                vtws[(((size_t)(b*HEADSc + head))*HDc + hd)*Nn + ntok] = (_Float16)acc[r];
            }
        }
    }
}

// ---------------------------------------------------------------------------
// Kernel 2: flash attention. 1 wave = 16 query rows, stream over 64-key blocks.
// V fragments are loaded BEFORE the softmax so the global loads complete under
// the VALU shadow; only the LDS P-bounce gates the PV WMMAs.
// ---------------------------------------------------------------------------
__global__ __launch_bounds__(128) void k_attn(
    const _Float16* __restrict__ qws, const _Float16* __restrict__ kws,
    const _Float16* __restrict__ vtws, _Float16* __restrict__ ows)
{
    __shared__ _Float16 sp[4][16][64];          // per-wave P tile (C-layout -> A-layout bounce)
    const int tid = threadIdx.x;
    const int wave = tid >> 5, lane = tid & 31;
    const int half = lane >> 4, lo = lane & 15;
    const int bh = blockIdx.y;                  // b*HEADS + head
    const int row0 = blockIdx.x*64 + wave*16;
    const float scale = 0.17677669529663687f;   // 1/sqrt(32)

    const _Float16* Q  = qws  + (size_t)bh*Nn*HDc;
    const _Float16* K  = kws  + (size_t)bh*Nn*HDc;
    const _Float16* Vt = vtws + (size_t)bh*HDc*Nn;

    v16h aq;                                    // Q fragment (HD=32 == one K-step), pre-scaled
    #pragma unroll
    for (int j = 0; j < 8; ++j) {
        int k0 = 16*(j>>2) + 8*half + 2*(j&3);
        H2 t; t.u = *(const unsigned int*)(Q + (size_t)(row0 + lo)*HDc + k0);
        aq[2*j]   = (_Float16)(scale * (float)t.h[0]);
        aq[2*j+1] = (_Float16)(scale * (float)t.h[1]);
    }

    v8f acc0 = {}, acc1 = {};                   // output accumulators: hd 0..15 / 16..31
    float mrow[8], lrow[8];
    #pragma unroll
    for (int r = 0; r < 8; ++r) { mrow[r] = -1e30f; lrow[r] = 0.f; }

    for (int kb = 0; kb < Nn; kb += 64) {
        v16h bk[4];                             // K block B-fragments (4 x 16 keys)
        #pragma unroll
        for (int t = 0; t < 4; ++t) {
            #pragma unroll
            for (int j = 0; j < 8; ++j) {
                int k0 = 16*(j>>2) + 8*half + 2*(j&3);
                H2 u; u.u = *(const unsigned int*)(K + (size_t)(kb + 16*t + lo)*HDc + k0);
                bk[t][2*j] = u.h[0]; bk[t][2*j+1] = u.h[1];
            }
        }
        v8f z = {};
        v8f s[4];
        #pragma unroll
        for (int t = 0; t < 4; ++t) s[t] = wmma16(aq, bk[t], z);

        v16h bv00, bv01, bv10, bv11;            // V (transposed): issue loads early
        #pragma unroll
        for (int j = 0; j < 8; ++j) {
            int k0 = 16*(j>>2) + 8*half + 2*(j&3);
            H2 t;
            t.u = *(const unsigned int*)(Vt + (size_t)lo*Nn        + kb      + k0);
            bv00[2*j] = t.h[0]; bv00[2*j+1] = t.h[1];
            t.u = *(const unsigned int*)(Vt + (size_t)(16+lo)*Nn   + kb      + k0);
            bv01[2*j] = t.h[0]; bv01[2*j+1] = t.h[1];
            t.u = *(const unsigned int*)(Vt + (size_t)lo*Nn        + kb + 32 + k0);
            bv10[2*j] = t.h[0]; bv10[2*j+1] = t.h[1];
            t.u = *(const unsigned int*)(Vt + (size_t)(16+lo)*Nn   + kb + 32 + k0);
            bv11[2*j] = t.h[0]; bv11[2*j+1] = t.h[1];
        }

        #pragma unroll
        for (int r = 0; r < 8; ++r) {           // online softmax per row (row lives in 16 lanes)
            float v = fmaxf(fmaxf(s[0][r], s[1][r]), fmaxf(s[2][r], s[3][r]));
            #pragma unroll
            for (int off = 8; off > 0; off >>= 1) v = fmaxf(v, __shfl_xor(v, off, 16));
            float mnew = fmaxf(mrow[r], v);
            float corr = __expf(mrow[r] - mnew);
            float p[4]; float rs = 0.f;
            #pragma unroll
            for (int t = 0; t < 4; ++t) { p[t] = __expf(s[t][r] - mnew); rs += p[t]; }
            #pragma unroll
            for (int off = 8; off > 0; off >>= 1) rs += __shfl_xor(rs, off, 16);
            lrow[r] = lrow[r]*corr + rs;
            mrow[r] = mnew;
            acc0[r] *= corr; acc1[r] *= corr;
            #pragma unroll
            for (int t = 0; t < 4; ++t)
                sp[wave][r + 8*half][lo + 16*t] = (_Float16)p[t];
        }
        asm volatile("s_wait_dscnt 0" ::: "memory");   // wave-local LDS bounce complete

        v16h ap0, ap1;                          // P reloaded in A-fragment layout (2 K-steps)
        const unsigned int* prow = (const unsigned int*)&sp[wave][lo][0];
        #pragma unroll
        for (int j = 0; j < 8; ++j) {
            int k0 = 16*(j>>2) + 8*half + 2*(j&3);
            H2 t0; t0.u = prow[k0 >> 1];
            ap0[2*j] = t0.h[0]; ap0[2*j+1] = t0.h[1];
            H2 t1; t1.u = prow[(k0 + 32) >> 1];
            ap1[2*j] = t1.h[0]; ap1[2*j+1] = t1.h[1];
        }
        acc0 = wmma16(ap0, bv00, acc0);
        acc0 = wmma16(ap1, bv10, acc0);
        acc1 = wmma16(ap0, bv01, acc1);
        acc1 = wmma16(ap1, bv11, acc1);
    }

    int b = bh >> 2, head = bh & 3;
    #pragma unroll
    for (int r = 0; r < 8; ++r) {
        int ntok = row0 + r + 8*half;
        float inv = 1.f / lrow[r];
        size_t base = ((size_t)(b*Nn + ntok))*Cc + head*HDc;
        ows[base + lo]      = (_Float16)(acc0[r]*inv);
        ows[base + 16 + lo] = (_Float16)(acc1[r]*inv);
    }
}

// ---------------------------------------------------------------------------
// Kernel 3: proj GEMM [16384x128]x[128x128] + bias + residual -> x1 [token][C] f32
// ---------------------------------------------------------------------------
__global__ __launch_bounds__(256) void k_proj(
    const _Float16* __restrict__ ows, const _Float16* __restrict__ projw,
    const float* __restrict__ projb, const float* __restrict__ x, float* __restrict__ x1)
{
    const int tid = threadIdx.x, blk = blockIdx.x;
    const int wave = tid >> 5, lane = tid & 31;
    const int half = lane >> 4, lo = lane & 15;
    const int rt = wave*16;
    const int trow = blk*128 + rt + lo;

    v16h afr[4];
    const unsigned int* arow = (const unsigned int*)(ows + (size_t)trow*Cc);
    #pragma unroll
    for (int s = 0; s < 4; ++s) {
        #pragma unroll
        for (int j = 0; j < 8; ++j) {
            int k0 = 32*s + 16*(j>>2) + 8*half + 2*(j&3);
            H2 t; t.u = arow[k0 >> 1];
            afr[s][2*j] = t.h[0]; afr[s][2*j+1] = t.h[1];
        }
    }
    for (int nt = 0; nt < 8; ++nt) {
        int n = nt*16 + lo;
        v16h bf[4];
        #pragma unroll
        for (int s = 0; s < 4; ++s) {
            #pragma unroll
            for (int j = 0; j < 8; ++j) {
                int k0 = 32*s + 16*(j>>2) + 8*half + 2*(j&3);
                H2 t; t.u = *(const unsigned int*)(projw + (size_t)n*Cc + k0);
                bf[s][2*j] = t.h[0]; bf[s][2*j+1] = t.h[1];
            }
        }
        v8f acc = {};
        #pragma unroll
        for (int s = 0; s < 4; ++s) acc = wmma16(afr[s], bf[s], acc);

        float bias = projb[n];
        #pragma unroll
        for (int r = 0; r < 8; ++r) {
            int token = blk*128 + rt + r + 8*half;
            int b = token >> 12, hw = token & 4095;
            float res = x[((size_t)(b*Cc + n))*Nn + hw];
            x1[(size_t)token*Cc + n] = acc[r] + bias + res;
        }
    }
}

// ---------------------------------------------------------------------------
// Kernel 4: LN2 + fc1 GEMM [16384x128]x[128x256] + exact GELU -> h f16 [token][256]
// ---------------------------------------------------------------------------
__global__ __launch_bounds__(256) void k_ln2_fc1(
    const float* __restrict__ x1, const float* __restrict__ ln2w, const float* __restrict__ ln2b,
    const _Float16* __restrict__ fc1w, const float* __restrict__ fc1b, _Float16* __restrict__ hws)
{
    __shared__ _Float16 sx[128][128];
    const int tid = threadIdx.x, blk = blockIdx.x;

    if (tid < 128) {
        int token = blk*128 + tid;
        const float* row = x1 + (size_t)token*Cc;
        float s = 0.f, ss = 0.f;
        for (int c = 0; c < Cc; ++c) { float v = row[c]; s += v; ss += v*v; }
        float mu = s * (1.f/128.f);
        float var = ss * (1.f/128.f) - mu*mu;
        float rstd = rsqrtf(var + 1e-5f);
        for (int c = 0; c < Cc; ++c)
            sx[tid][c] = (_Float16)((row[c] - mu)*rstd*ln2w[c] + ln2b[c]);
    }
    __syncthreads();

    const int wave = tid >> 5, lane = tid & 31;
    const int half = lane >> 4, lo = lane & 15;
    const int rt = wave*16;

    v16h afr[4];
    const unsigned int* srow = (const unsigned int*)&sx[rt + lo][0];
    #pragma unroll
    for (int s = 0; s < 4; ++s) {
        #pragma unroll
        for (int j = 0; j < 8; ++j) {
            int k0 = 32*s + 16*(j>>2) + 8*half + 2*(j&3);
            H2 t; t.u = srow[k0 >> 1];
            afr[s][2*j] = t.h[0]; afr[s][2*j+1] = t.h[1];
        }
    }
    for (int nt = 0; nt < 16; ++nt) {           // 256 outputs / 16
        int n = nt*16 + lo;
        v16h bf[4];
        #pragma unroll
        for (int s = 0; s < 4; ++s) {
            #pragma unroll
            for (int j = 0; j < 8; ++j) {
                int k0 = 32*s + 16*(j>>2) + 8*half + 2*(j&3);
                H2 t; t.u = *(const unsigned int*)(fc1w + (size_t)n*Cc + k0);
                bf[s][2*j] = t.h[0]; bf[s][2*j+1] = t.h[1];
            }
        }
        v8f acc = {};
        #pragma unroll
        for (int s = 0; s < 4; ++s) acc = wmma16(afr[s], bf[s], acc);

        float bias = fc1b[n];
        #pragma unroll
        for (int r = 0; r < 8; ++r) {
            int token = blk*128 + rt + r + 8*half;
            float v = acc[r] + bias;
            float g = 0.5f * v * (1.f + erff(v * 0.70710678118654752f));
            hws[(size_t)token*HIDDENc + n] = (_Float16)g;
        }
    }
}

// ---------------------------------------------------------------------------
// Kernel 5: fc2 GEMM [16384x256]x[256x128] + bias + residual -> d_out NCHW f32
// ---------------------------------------------------------------------------
__global__ __launch_bounds__(256) void k_fc2(
    const _Float16* __restrict__ hws, const _Float16* __restrict__ fc2w,
    const float* __restrict__ fc2b, const float* __restrict__ x1, float* __restrict__ out)
{
    const int tid = threadIdx.x, blk = blockIdx.x;
    const int wave = tid >> 5, lane = tid & 31;
    const int half = lane >> 4, lo = lane & 15;
    const int rt = wave*16;
    const int trow = blk*128 + rt + lo;

    v16h afr[8];                                // K = 256 -> 8 K-steps
    const unsigned int* arow = (const unsigned int*)(hws + (size_t)trow*HIDDENc);
    #pragma unroll
    for (int s = 0; s < 8; ++s) {
        #pragma unroll
        for (int j = 0; j < 8; ++j) {
            int k0 = 32*s + 16*(j>>2) + 8*half + 2*(j&3);
            H2 t; t.u = arow[k0 >> 1];
            afr[s][2*j] = t.h[0]; afr[s][2*j+1] = t.h[1];
        }
    }
    for (int nt = 0; nt < 8; ++nt) {
        int n = nt*16 + lo;
        v8f acc = {};
        #pragma unroll
        for (int g = 0; g < 2; ++g) {           // 2 groups of 4 K-steps (VGPR budget)
            v16h bf[4];
            #pragma unroll
            for (int s = 0; s < 4; ++s) {
                #pragma unroll
                for (int j = 0; j < 8; ++j) {
                    int k0 = 32*(4*g + s) + 16*(j>>2) + 8*half + 2*(j&3);
                    H2 t; t.u = *(const unsigned int*)(fc2w + (size_t)n*HIDDENc + k0);
                    bf[s][2*j] = t.h[0]; bf[s][2*j+1] = t.h[1];
                }
            }
            #pragma unroll
            for (int s = 0; s < 4; ++s) acc = wmma16(afr[4*g + s], bf[s], acc);
        }
        float bias = fc2b[n];
        #pragma unroll
        for (int r = 0; r < 8; ++r) {
            int token = blk*128 + rt + r + 8*half;
            int b = token >> 12, hw = token & 4095;
            out[((size_t)(b*Cc + n))*Nn + hw] = acc[r] + bias + x1[(size_t)token*Cc + n];
        }
    }
}

// ---------------------------------------------------------------------------
extern "C" void kernel_launch(void* const* d_in, const int* in_sizes, int n_in,
                              void* d_out, int out_size, void* d_ws, size_t ws_size,
                              hipStream_t stream) {
    const float* x     = (const float*)d_in[0];
    const float* ln1w  = (const float*)d_in[1];
    const float* ln1b  = (const float*)d_in[2];
    const float* qkvw  = (const float*)d_in[3];
    const float* projw = (const float*)d_in[4];
    const float* projb = (const float*)d_in[5];
    const float* ln2w  = (const float*)d_in[6];
    const float* ln2b  = (const float*)d_in[7];
    const float* fc1w  = (const float*)d_in[8];
    const float* fc1b  = (const float*)d_in[9];
    const float* fc2w  = (const float*)d_in[10];
    const float* fc2b  = (const float*)d_in[11];

    char* ws = (char*)d_ws;
    _Float16* qws   = (_Float16*)(ws);                     //  4 MB  Q  f16 [bh][n][32]
    _Float16* kws   = (_Float16*)(ws + ( 4u << 20));       //  4 MB  K  f16 [bh][n][32]
    _Float16* vtws  = (_Float16*)(ws + ( 8u << 20));       //  4 MB  Vt f16 [bh][32][n]
    _Float16* owsb  = (_Float16*)(ws + (12u << 20));       //  4 MB  attn out f16 [token][C]
    float*    x1    = (float*)   (ws + (16u << 20));       //  8 MB  residual f32 [token][C]
    _Float16* hws   = (_Float16*)(ws + (24u << 20));       //  8 MB  MLP hidden f16 [token][256]
    _Float16* wqkvh = (_Float16*)(ws + (32u << 20));       //  96 KB qkv_w  f16
    _Float16* wprjh = (_Float16*)(ws + (33u << 20));       //  32 KB proj_w f16
    _Float16* wfc1h = (_Float16*)(ws + (34u << 20));       //  64 KB fc1_w  f16
    _Float16* wfc2h = (_Float16*)(ws + (35u << 20));       //  64 KB fc2_w  f16
    float* out = (float*)d_out;

    // one-shot weight conversion (tiny; L2-resident afterwards)
    k_cvt<<<dim3((3*Cc*Cc + 255)/256), 256, 0, stream>>>(qkvw, wqkvh, 3*Cc*Cc);
    k_cvt<<<dim3((Cc*Cc + 255)/256), 256, 0, stream>>>(projw, wprjh, Cc*Cc);
    k_cvt<<<dim3((HIDDENc*Cc + 255)/256), 256, 0, stream>>>(fc1w, wfc1h, HIDDENc*Cc);
    k_cvt<<<dim3((Cc*HIDDENc + 255)/256), 256, 0, stream>>>(fc2w, wfc2h, Cc*HIDDENc);

    k_ln_qkv <<<dim3(NTOK/128), 256, 0, stream>>>(x, ln1w, ln1b, wqkvh, qws, kws, vtws);
    k_attn   <<<dim3(Nn/64, Bc*HEADSc), 128, 0, stream>>>(qws, kws, vtws, owsb);
    k_proj   <<<dim3(NTOK/128), 256, 0, stream>>>(owsb, wprjh, projb, x, x1);
    k_ln2_fc1<<<dim3(NTOK/128), 256, 0, stream>>>(x1, ln2w, ln2b, wfc1h, fc1b, hws);
    k_fc2    <<<dim3(NTOK/128), 256, 0, stream>>>(hws, wfc2h, fc2b, x1, out);
}
